// MultiHeadAttention_77549929496957
// MI455X (gfx1250) — matrix operand order repaired and checked by reading, not compiled, not run on hardware
//
#include <hip/hip_runtime.h>
#include <stdint.h>

#define DM    1024
#define HEADS 16
#define DK    64
#define BATCH 8
#define SEQ   1024

#define BLK_M 128
#define BLK_N 64
#define BLK_K 32

typedef __attribute__((ext_vector_type(16))) __bf16 v16bf;
typedef __attribute__((ext_vector_type(8)))  __bf16 v8bf;
typedef __attribute__((ext_vector_type(8)))  float  v8f;

__device__ __forceinline__ unsigned short f2bf(float f) {
  unsigned int u = __builtin_bit_cast(unsigned int, f);
  u += 0x7FFFu + ((u >> 16) & 1u);   // round-to-nearest-even
  return (unsigned short)(u >> 16);
}

__device__ __forceinline__ v16bf ld_v16(const __bf16* p0, const __bf16* p1) {
  v8bf lo = *reinterpret_cast<const v8bf*>(p0);
  v8bf hv = *reinterpret_cast<const v8bf*>(p1);
  v16bf r;
#pragma unroll
  for (int i = 0; i < 8; ++i) { r[i] = lo[i]; r[i + 8] = hv[i]; }
  return r;
}

// LDS byte offset of a generic pointer into shared memory (addrspace(3) offset).
__device__ __forceinline__ unsigned lds_off(const void* p) {
  return (unsigned)(uintptr_t)(const __attribute__((address_space(3))) void*)p;
}

// 16-byte per-lane async copy global -> LDS (ASYNCcnt-tracked, no VGPR data).
__device__ __forceinline__ void async_copy16(unsigned lds_byte, const void* gsrc) {
  unsigned long long ga = (unsigned long long)(uintptr_t)gsrc;
  asm volatile("global_load_async_to_lds_b128 %0, %1, off"
               :: "v"(lds_byte), "v"(ga)
               : "memory");
}

// ---------------------------------------------------------------- fp32 -> bf16
__global__ void cvt_f32_to_bf16_kernel(const float* __restrict__ in,
                                       unsigned short* __restrict__ out, int n4) {
  int i = blockIdx.x * blockDim.x + threadIdx.x;
  if (i < n4) {
    float4 f = reinterpret_cast<const float4*>(in)[i];
    ushort4 o;
    o.x = f2bf(f.x); o.y = f2bf(f.y); o.z = f2bf(f.z); o.w = f2bf(f.w);
    reinterpret_cast<ushort4*>(out)[i] = o;
  }
}

// ---------------------------------------------------------------- GEMM: Y = A @ B^T
// A: [M x K] bf16 row-major.  B: [N x K] bf16 row-major (torch weight layout).
// Block tile 128x64 staged in double-buffered LDS via async-to-LDS DMA;
// 8 waves, each wave owns a 32x32 register tile (4 WMMA accumulators).
// MODE 0: Out fp32 [M x N].  MODE 1: Out bf16 [B,H,S,DK].  MODE 2: Out bf16 [B,H,DK,S].
template <int MODE>
__global__ __launch_bounds__(256)
void gemm_bf16_wmma(const __bf16* __restrict__ A, const __bf16* __restrict__ B,
                    void* __restrict__ Out, int M, int N, int K) {
  __shared__ __align__(16) unsigned short smem[2 * (BLK_M * BLK_K + BLK_N * BLK_K)];

  const int t     = threadIdx.x;
  const int lane  = t & 31;
  const int wave  = t >> 5;
  const int r     = lane & 15;
  const int hi    = lane >> 4;
  const int waveM = wave & 3;    // 4 M-tiles of 32
  const int waveN = wave >> 2;   // 2 N-tiles of 32
  const int mBase = blockIdx.x * BLK_M;
  const int nBase = blockIdx.y * BLK_N;

  const unsigned bufStride = BLK_M * BLK_K + BLK_N * BLK_K;  // elems (6144)
  const unsigned bOffElem  = BLK_M * BLK_K;                  // elems (4096)
  const __bf16*  sm        = reinterpret_cast<const __bf16*>(smem);

  // Stage one K-step block tile into LDS buffer `buf` (3 async ops / thread;
  // 512 A-chunks + 256 B-chunks of 16B == exactly 3*256, no divergence).
  auto stage = [&](int buf, int kk) {
    const unsigned base = (unsigned)buf * bufStride;
#pragma unroll
    for (int i = 0; i < 2; ++i) {
      const int c = t + 256 * i;             // 0..511
      const int row = c >> 2, col8 = c & 3;  // [128 rows][4 x 8-elem chunks]
      async_copy16(lds_off(&smem[base + row * BLK_K + col8 * 8]),
                   A + (size_t)(mBase + row) * K + kk + col8 * 8);
    }
    {
      const int row = t >> 2, col8 = t & 3;  // [64 rows][4 chunks]
      async_copy16(lds_off(&smem[base + bOffElem + row * BLK_K + col8 * 8]),
                   B + (size_t)(nBase + row) * K + kk + col8 * 8);
    }
  };

  v8f c[2][2] = {{v8f{}, v8f{}}, {v8f{}, v8f{}}};

  int buf = 0;
  stage(0, 0);
  for (int kk = 0; kk < K; kk += BLK_K) {
    if (kk + BLK_K < K) {
      stage(buf ^ 1, kk + BLK_K);                       // prefetch next tile
      asm volatile("s_wait_asynccnt 3" ::: "memory");   // prior stage done (in-order)
    } else {
      asm volatile("s_wait_asynccnt 0" ::: "memory");
    }
    __syncthreads();

    const unsigned base = (unsigned)buf * bufStride;
    const int ar0 = waveM * 32 + r;        // A frag rows (lane = M row)
    const int ar1 = ar0 + 16;
    const int bc0 = waveN * 32 + r;        // B frag cols (lane = N col)
    const int bc1 = bc0 + 16;
    v16bf a0 = ld_v16(sm + base + ar0 * BLK_K + hi * 8,
                      sm + base + ar0 * BLK_K + 16 + hi * 8);
    v16bf a1 = ld_v16(sm + base + ar1 * BLK_K + hi * 8,
                      sm + base + ar1 * BLK_K + 16 + hi * 8);
    v16bf b0 = ld_v16(sm + base + bOffElem + bc0 * BLK_K + hi * 8,
                      sm + base + bOffElem + bc0 * BLK_K + 16 + hi * 8);
    v16bf b1 = ld_v16(sm + base + bOffElem + bc1 * BLK_K + hi * 8,
                      sm + base + bOffElem + bc1 * BLK_K + 16 + hi * 8);

    c[0][0] = __builtin_amdgcn_wmma_f32_16x16x32_bf16(false, a0, false, b0, (short)0, c[0][0], false, false);
    c[0][1] = __builtin_amdgcn_wmma_f32_16x16x32_bf16(false, a0, false, b1, (short)0, c[0][1], false, false);
    c[1][0] = __builtin_amdgcn_wmma_f32_16x16x32_bf16(false, a1, false, b0, (short)0, c[1][0], false, false);
    c[1][1] = __builtin_amdgcn_wmma_f32_16x16x32_bf16(false, a1, false, b1, (short)0, c[1][1], false, false);

    __syncthreads();
    buf ^= 1;
  }

#pragma unroll
  for (int mf = 0; mf < 2; ++mf)
#pragma unroll
    for (int nf = 0; nf < 2; ++nf)
#pragma unroll
      for (int j = 0; j < 8; ++j) {
        const int m = mBase + waveM * 32 + mf * 16 + j + 8 * hi;
        const int n = nBase + waveN * 32 + nf * 16 + r;
        if (MODE == 0) {
          reinterpret_cast<float*>(Out)[(size_t)m * N + n] = c[mf][nf][j];
        } else {
          const int b_ = m >> 10, s = m & (SEQ - 1);
          const int h  = n >> 6,  d = n & (DK - 1);
          const unsigned short val = f2bf(c[mf][nf][j]);
          if (MODE == 1)
            reinterpret_cast<unsigned short*>(Out)
                [(((size_t)b_ * HEADS + h) * SEQ + s) * DK + d] = val;
          else
            reinterpret_cast<unsigned short*>(Out)
                [(((size_t)b_ * HEADS + h) * DK + d) * SEQ + s] = val;
        }
      }
}

// ---------------------------------------------------------------- attention
// q,k: bf16 [B*H, S, DK].  vT: bf16 [B*H, DK, S].  ctx out: bf16 [B, S, H*DK].
__global__ __launch_bounds__(32)
void attn_wmma(const __bf16* __restrict__ q, const __bf16* __restrict__ k,
               const __bf16* __restrict__ vT, unsigned short* __restrict__ ctx) {
  const int lane  = threadIdx.x & 31;
  const int r     = lane & 15;
  const int hi    = lane >> 4;
  const int bh    = blockIdx.y;
  const int qBase = blockIdx.x * 16;

  const __bf16* qRow = q + ((size_t)bh * SEQ + qBase + r) * DK;
  const v16bf qa0 = ld_v16(qRow + hi * 8,      qRow + 16 + hi * 8);  // d 0..31
  const v16bf qa1 = ld_v16(qRow + 32 + hi * 8, qRow + 48 + hi * 8);  // d 32..63

  float mrow[8], lrow[8];
  v8f o[4] = {v8f{}, v8f{}, v8f{}, v8f{}};
#pragma unroll
  for (int j = 0; j < 8; ++j) { mrow[j] = -3.0e38f; lrow[j] = 0.0f; }

  __shared__ __align__(16) unsigned short pLds[16 * 32];

  for (int kb = 0; kb < SEQ; kb += 32) {
    // ---- scores S = Q K^T * 1/sqrt(dk), two 16x16 fragments (32 keys)
    v8f s0 = {}, s1 = {};
    {
      const __bf16* kR0 = k + ((size_t)bh * SEQ + kb + r) * DK;
      const __bf16* kR1 = k + ((size_t)bh * SEQ + kb + 16 + r) * DK;
      v16bf k00 = ld_v16(kR0 + hi * 8,      kR0 + 16 + hi * 8);
      v16bf k01 = ld_v16(kR0 + 32 + hi * 8, kR0 + 48 + hi * 8);
      v16bf k10 = ld_v16(kR1 + hi * 8,      kR1 + 16 + hi * 8);
      v16bf k11 = ld_v16(kR1 + 32 + hi * 8, kR1 + 48 + hi * 8);
      s0 = __builtin_amdgcn_wmma_f32_16x16x32_bf16(false, qa0, false, k00, (short)0, s0, false, false);
      s0 = __builtin_amdgcn_wmma_f32_16x16x32_bf16(false, qa1, false, k01, (short)0, s0, false, false);
      s1 = __builtin_amdgcn_wmma_f32_16x16x32_bf16(false, qa0, false, k10, (short)0, s1, false, false);
      s1 = __builtin_amdgcn_wmma_f32_16x16x32_bf16(false, qa1, false, k11, (short)0, s1, false, false);
    }

    // ---- online softmax update (row = j + 8*hi, spread over 16 lanes)
#pragma unroll
    for (int j = 0; j < 8; ++j) {
      float a0 = s0[j] * 0.125f;
      float a1 = s1[j] * 0.125f;
      float rm = fmaxf(a0, a1);
      rm = fmaxf(rm, __shfl_xor(rm, 1, 16));
      rm = fmaxf(rm, __shfl_xor(rm, 2, 16));
      rm = fmaxf(rm, __shfl_xor(rm, 4, 16));
      rm = fmaxf(rm, __shfl_xor(rm, 8, 16));
      float mnew  = fmaxf(mrow[j], rm);
      float alpha = __expf(mrow[j] - mnew);
      float p0 = __expf(a0 - mnew);
      float p1 = __expf(a1 - mnew);
      float rs = p0 + p1;
      rs += __shfl_xor(rs, 1, 16);
      rs += __shfl_xor(rs, 2, 16);
      rs += __shfl_xor(rs, 4, 16);
      rs += __shfl_xor(rs, 8, 16);
      lrow[j] = lrow[j] * alpha + rs;
      mrow[j] = mnew;
      o[0][j] *= alpha; o[1][j] *= alpha; o[2][j] *= alpha; o[3][j] *= alpha;
      const int row = j + 8 * hi;
      pLds[row * 32 + r]      = f2bf(p0);
      pLds[row * 32 + 16 + r] = f2bf(p1);
    }
    __syncthreads();   // single wave: LDS ops in-order; folds to s_nop

    // ---- re-pack P from C-layout to A-layout via LDS
    const v16bf pa =
        ld_v16(reinterpret_cast<const __bf16*>(&pLds[r * 32 + hi * 8]),
               reinterpret_cast<const __bf16*>(&pLds[r * 32 + 16 + hi * 8]));

    // ---- O += P @ V (V^T rows are key-contiguous)
#pragma unroll
    for (int tt = 0; tt < 4; ++tt) {
      const __bf16* vRow = vT + ((size_t)bh * DK + tt * 16 + r) * SEQ + kb;
      v16bf vb = ld_v16(vRow + hi * 8, vRow + 16 + hi * 8);
      o[tt] = __builtin_amdgcn_wmma_f32_16x16x32_bf16(false, pa, false, vb,
                                                      (short)0, o[tt], false, false);
    }
    __syncthreads();
  }

  // ---- normalize and write context [B, S, H*DK]
  const int b_ = bh / HEADS, h = bh % HEADS;
#pragma unroll
  for (int j = 0; j < 8; ++j) {
    const float inv = 1.0f / lrow[j];
    const int   row = qBase + j + 8 * hi;
    const size_t base = ((size_t)b_ * SEQ + row) * DM + h * DK;
#pragma unroll
    for (int tt = 0; tt < 4; ++tt)
      ctx[base + tt * 16 + r] = f2bf(o[tt][j] * inv);
  }
}

// ---------------------------------------------------------------- launch
extern "C" void kernel_launch(void* const* d_in, const int* in_sizes, int n_in,
                              void* d_out, int out_size, void* d_ws, size_t ws_size,
                              hipStream_t stream) {
  const float* Q   = (const float*)d_in[0];
  const float* K   = (const float*)d_in[1];
  const float* V   = (const float*)d_in[2];
  const float* WQ  = (const float*)d_in[3];
  const float* WK  = (const float*)d_in[4];
  const float* WV  = (const float*)d_in[5];
  const float* Wfc = (const float*)d_in[6];

  const size_t nX = (size_t)BATCH * SEQ * DM;   // 8M elems
  const size_t nW = (size_t)DM * DM;            // 1M elems

  unsigned short* bX   = (unsigned short*)d_ws;  // staged bf16 input (reused; aliased by ctx)
  unsigned short* bWQ  = bX + nX;
  unsigned short* bWK  = bWQ + nW;
  unsigned short* bWV  = bWK + nW;
  unsigned short* bWfc = bWV + nW;
  unsigned short* qB   = bWfc + nW;              // [B,H,S,DK]
  unsigned short* kB   = qB + nX;                // [B,H,S,DK]
  unsigned short* vTB  = kB + nX;                // [B,H,DK,S]
  unsigned short* ctx  = bX;                     // alias: input staging dead by then

  auto cvt = [&](const float* in, unsigned short* out, size_t n) {
    int n4 = (int)(n / 4);
    cvt_f32_to_bf16_kernel<<<dim3((n4 + 255) / 256), dim3(256), 0, stream>>>(in, out, n4);
  };

  cvt(WQ, bWQ, nW); cvt(WK, bWK, nW); cvt(WV, bWV, nW); cvt(Wfc, bWfc, nW);

  const dim3 gemmGrid(BATCH * SEQ / BLK_M, DM / BLK_N);  // (64, 16)
  const dim3 gemmBlk(256);

  cvt(Q, bX, nX);
  gemm_bf16_wmma<1><<<gemmGrid, gemmBlk, 0, stream>>>(
      (const __bf16*)bX, (const __bf16*)bWQ, qB, BATCH * SEQ, DM, DM);
  cvt(K, bX, nX);
  gemm_bf16_wmma<1><<<gemmGrid, gemmBlk, 0, stream>>>(
      (const __bf16*)bX, (const __bf16*)bWK, kB, BATCH * SEQ, DM, DM);
  cvt(V, bX, nX);
  gemm_bf16_wmma<2><<<gemmGrid, gemmBlk, 0, stream>>>(
      (const __bf16*)bX, (const __bf16*)bWV, vTB, BATCH * SEQ, DM, DM);

  const dim3 attnGrid(SEQ / 16, BATCH * HEADS);          // (64, 128)
  attn_wmma<<<attnGrid, dim3(32), 0, stream>>>(
      (const __bf16*)qB, (const __bf16*)kB, (const __bf16*)vTB, ctx);

  gemm_bf16_wmma<0><<<gemmGrid, gemmBlk, 0, stream>>>(
      (const __bf16*)ctx, (const __bf16*)bWfc, d_out, BATCH * SEQ, DM, DM);
}
